// LinearMemoryAttention_11218454577699
// MI455X (gfx1250) — compile-verified, matrix operand order checked by reading, not run on hardware
//
#include <hip/hip_runtime.h>
#include <hip/hip_bf16.h>

// Problem constants (from the reference)
#define HIDDEN  1024
#define NHEADS  16
#define HD      64
#define BATCH   4
#define SEQ     8192
#define MROWS   (BATCH * SEQ)      // 32768
#define LMA_EPS 1e-6f

typedef __bf16 bf16_t;
typedef __attribute__((ext_vector_type(16))) __bf16 v16bf;
typedef __attribute__((ext_vector_type(8)))  __bf16 v8bf;
typedef __attribute__((ext_vector_type(8)))  float  v8f;
typedef __attribute__((ext_vector_type(4)))  int    i32x4;

// Address-space-qualified 16-byte vector types for the async-copy builtin
// (clang's gfx1250 builtin expects v4i32 pointees; AS1 = global, AS3 = LDS).
typedef __attribute__((address_space(1))) i32x4 i32x4_glob;
typedef __attribute__((address_space(3))) i32x4 i32x4_lds;

union FragAB { v16bf v; v8bf h[2]; };

// ---------------------------------------------------------------------------
// Async global->LDS copy (GLOBAL_LOAD_ASYNC_TO_LDS_B128, ASYNCcnt-tracked).
// Falls back to a synchronous VGPR round-trip if the builtin is unavailable.
// ---------------------------------------------------------------------------
__device__ __forceinline__ void async_copy16(const void* g, void* l) {
#if __has_builtin(__builtin_amdgcn_global_load_async_to_lds_b128)
  __builtin_amdgcn_global_load_async_to_lds_b128(
      (i32x4_glob*)g, (i32x4_lds*)l, 0, 0);
#else
  *(v8bf*)l = *(const v8bf*)g;
#endif
}

__device__ __forceinline__ void async_wait_all() {
#if __has_builtin(__builtin_amdgcn_s_wait_asynccnt)
  __builtin_amdgcn_s_wait_asynccnt(0);
#elif __has_builtin(__builtin_amdgcn_global_load_async_to_lds_b128)
  asm volatile("s_wait_asynccnt 0x0" ::: "memory");
#endif
}

// ---------------------------------------------------------------------------
// Conversion kernels (one cheap pass each; bf16 halves all later tile traffic)
// ---------------------------------------------------------------------------
__global__ __launch_bounds__(256)
void lma_cvt_f32_bf16_kernel(const float* __restrict__ src,
                             bf16_t* __restrict__ dst, int n) {
  int i = blockIdx.x * blockDim.x + threadIdx.x;
  if (i < n) dst[i] = (bf16_t)src[i];
}

// Vectorized X conversion: 8 elements / thread, float4 loads, b128 store.
__global__ __launch_bounds__(256)
void lma_cvt_x_kernel(const float* __restrict__ src, bf16_t* __restrict__ dst) {
  const size_t i = ((size_t)blockIdx.x * 256 + threadIdx.x) * 8;
  const float4 f0 = *(const float4*)(src + i);
  const float4 f1 = *(const float4*)(src + i + 4);
  v8bf o;
  o[0] = (bf16_t)f0.x; o[1] = (bf16_t)f0.y; o[2] = (bf16_t)f0.z; o[3] = (bf16_t)f0.w;
  o[4] = (bf16_t)f1.x; o[5] = (bf16_t)f1.y; o[6] = (bf16_t)f1.z; o[7] = (bf16_t)f1.w;
  *(v8bf*)(dst + i) = o;
}

// S0 is [bh][d][e]; retrieve-GEMM wants B^T layout [bh][e][d] in bf16.
__global__ __launch_bounds__(256)
void lma_s0_transpose_kernel(const float* __restrict__ S0,
                             bf16_t* __restrict__ S0t) {
  const int bh = blockIdx.x;
  const float* src = S0 + (size_t)bh * (HD * HD);
  bf16_t*      dst = S0t + (size_t)bh * (HD * HD);
  for (int i = threadIdx.x; i < HD * HD; i += blockDim.x) {
    const int d = i >> 6, e = i & 63;
    dst[e * HD + d] = (bf16_t)src[d * HD + e];
  }
}

// ---------------------------------------------------------------------------
// Main GEMM:  C[M,N] = A[M,1024] * W^T, A bf16, W row-major [N,K] (== B^T
// layout for the WMMA B-fragment loader).  All tile staging is async
// global->LDS (double buffered); compute is v_wmma_f32_16x16x32_bf16.
//
// MODE 0: epilogue = +bias, phi = elu()+1, bf16 head-major store (for retrieve)
// MODE 1: epilogue = fp32 store to d_out
//
// Block tile 128x128, BK=32, 256 threads = 8 wave32, wave tile 32x64
// -> 8 WMMA per k-step, 32 k-steps = 256 WMMA issues per wave.
// ---------------------------------------------------------------------------
template <int MODE>
__global__ __launch_bounds__(256)
void lma_gemm_kernel(const bf16_t* __restrict__ Ab, const bf16_t* __restrict__ Wb,
                     const float* __restrict__ bias, void* __restrict__ Cg_) {
  __shared__ bf16_t As[2][128][32];
  __shared__ bf16_t Bs[2][128][32];

  const int tid    = threadIdx.x;
  const int lane   = tid & 31;
  const int wave   = tid >> 5;
  const int lane_r = lane & 15;
  const int hi     = lane >> 4;        // 0: lanes 0-15, 1: lanes 16-31

  const int wm = wave & 3;             // 4 waves along M -> 32-row slices
  const int wn = wave >> 2;            // 2 waves along N -> 64-col slices
  const int n0 = blockIdx.x * 128;
  const int m0 = blockIdx.y * 128;

  v8f acc[2][4];
#pragma unroll
  for (int i = 0; i < 2; ++i)
#pragma unroll
    for (int j = 0; j < 4; ++j)
#pragma unroll
      for (int r = 0; r < 8; ++r) acc[i][j][r] = 0.0f;

  // Stage one 128x32 bf16 A tile + B tile into LDS via async b128 copies.
  // 4096 bf16 per tile = 512 x 16B chunks -> 2 chunks/thread/tile.
  auto stage_tiles = [&](int kt, int buf) {
    const int k0 = kt * 32;
#pragma unroll
    for (int i = 0; i < 2; ++i) {
      const int t   = tid + i * 256;
      const int row = t >> 2;
      const int c8  = (t & 3) << 3;
      async_copy16(Ab + (size_t)(m0 + row) * HIDDEN + k0 + c8,
                   &As[buf][row][c8]);
      async_copy16(Wb + (size_t)(n0 + row) * HIDDEN + k0 + c8,
                   &Bs[buf][row][c8]);
    }
    // Prefetch the tile after next into GL2 (global_prefetch_b8)
    const int kn = k0 + 64;
    if (kn < HIDDEN) {
      __builtin_prefetch(Wb + (size_t)(n0 + (tid >> 1)) * HIDDEN + kn, 0, 0);
    }
  };

  stage_tiles(0, 0);
  async_wait_all();
  __syncthreads();

  const int KT = HIDDEN / 32;  // 32 k-steps
  for (int kt = 0; kt < KT; ++kt) {
    const int buf = kt & 1;
    if (kt + 1 < KT) stage_tiles(kt + 1, buf ^ 1);  // async, overlaps compute

    // A fragments: lane_r = M row, halves K in {kb..kb+7, 16+kb..16+kb+7}
    const int kb = hi * 8;
    FragAB afr[2], bfr[4];
#pragma unroll
    for (int i = 0; i < 2; ++i) {
      const int row = wm * 32 + i * 16 + lane_r;
      afr[i].h[0] = *(const v8bf*)&As[buf][row][kb];
      afr[i].h[1] = *(const v8bf*)&As[buf][row][16 + kb];
    }
#pragma unroll
    for (int j = 0; j < 4; ++j) {
      const int row = wn * 64 + j * 16 + lane_r;
      bfr[j].h[0] = *(const v8bf*)&Bs[buf][row][kb];
      bfr[j].h[1] = *(const v8bf*)&Bs[buf][row][16 + kb];
    }
#pragma unroll
    for (int i = 0; i < 2; ++i)
#pragma unroll
      for (int j = 0; j < 4; ++j)
        acc[i][j] = __builtin_amdgcn_wmma_f32_16x16x32_bf16(
            false, afr[i].v, false, bfr[j].v, (short)0, acc[i][j], false, false);

    async_wait_all();   // this wave's copies into buf^1 have landed in LDS
    __syncthreads();    // all waves' copies have landed
  }

  // Epilogue (C/D layout: VGPR r -> M = r (lanes 0-15) / 8+r (lanes 16-31))
#pragma unroll
  for (int i = 0; i < 2; ++i) {
#pragma unroll
    for (int j = 0; j < 4; ++j) {
#pragma unroll
      for (int r = 0; r < 8; ++r) {
        const int gm = m0 + wm * 32 + i * 16 + hi * 8 + r;
        const int gn = n0 + wn * 64 + j * 16 + lane_r;
        float v = acc[i][j][r];
        if constexpr (MODE == 0) {
          v += bias[gn];
          v = (v > 0.0f) ? (v + 1.0f) : __expf(v);  // phi = elu(x)+1
          const int hh = gn >> 6, dd = gn & 63;
          const int bb = gm >> 13, ss = gm & (SEQ - 1);
          ((bf16_t*)Cg_)[(((size_t)(bb * NHEADS + hh)) * SEQ + ss) * HD + dd] =
              (bf16_t)v;
        } else {
          ((float*)Cg_)[(size_t)gm * HIDDEN + gn] = v;
        }
      }
    }
  }
}

// ---------------------------------------------------------------------------
// Retrieve: per (b,h): att[s,e] = (qf[s,:] @ S0[:, e]) * rcp(qf[s,:].z0 + eps)
// Block = 128 rows x 64 cols, K=64 -> 8 WMMA per wave; one v_rcp per row.
// Output bf16, merged-head layout [m, h*64+e] for the O-projection.
// ---------------------------------------------------------------------------
__global__ __launch_bounds__(256)
void lma_retrieve_kernel(const bf16_t* __restrict__ qf,
                         const bf16_t* __restrict__ S0t,
                         const float* __restrict__ z0,
                         bf16_t* __restrict__ att) {
  __shared__ bf16_t qs[128][64];
  __shared__ bf16_t Ss[64][64];
  __shared__ float  rden[128];

  const int tid    = threadIdx.x;
  const int lane   = tid & 31;
  const int wave   = tid >> 5;         // 8 waves -> one 16-row M tile each
  const int lane_r = lane & 15;
  const int hi     = lane >> 4;

  const int bh = blockIdx.y;           // b*16 + h
  const int b  = bh >> 4;
  const int h  = bh & 15;
  const int s0 = blockIdx.x * 128;

  const bf16_t* qbase = qf + ((size_t)bh * SEQ + s0) * HD;
#pragma unroll
  for (int i = 0; i < 4; ++i) {
    const int t = tid + i * 256;
    const int row = t >> 3, c8 = (t & 7) << 3;
    async_copy16(qbase + (size_t)row * HD + c8, &qs[row][c8]);
  }
#pragma unroll
  for (int i = 0; i < 2; ++i) {
    const int t = tid + i * 256;
    const int row = t >> 3, c8 = (t & 7) << 3;
    async_copy16(S0t + (size_t)bh * (HD * HD) + (size_t)row * HD + c8,
                 &Ss[row][c8]);
  }
  async_wait_all();
  __syncthreads();

  // rden[s] = 1 / (eps + qf[s,:] . z0[bh,:])  -- one v_rcp_f32 per row;
  // denominator is positive and well-conditioned (phi>=0, z0>=0.1).
  if (tid < 128) {
    const float* zp = z0 + (size_t)bh * HD;
    float s = LMA_EPS;
#pragma unroll 8
    for (int d = 0; d < HD; ++d) s += (float)qs[tid][d] * zp[d];
    rden[tid] = __builtin_amdgcn_rcpf(s);
  }

  v8f acc[4];
#pragma unroll
  for (int j = 0; j < 4; ++j)
#pragma unroll
    for (int r = 0; r < 8; ++r) acc[j][r] = 0.0f;

  const int kb = hi * 8;
#pragma unroll
  for (int ks = 0; ks < 2; ++ks) {
    const int k0 = ks * 32;
    FragAB a;
    const int arow = wave * 16 + lane_r;
    a.h[0] = *(const v8bf*)&qs[arow][k0 + kb];
    a.h[1] = *(const v8bf*)&qs[arow][k0 + 16 + kb];
#pragma unroll
    for (int j = 0; j < 4; ++j) {
      FragAB bb2;
      const int brow = j * 16 + lane_r;
      bb2.h[0] = *(const v8bf*)&Ss[brow][k0 + kb];
      bb2.h[1] = *(const v8bf*)&Ss[brow][k0 + 16 + kb];
      acc[j] = __builtin_amdgcn_wmma_f32_16x16x32_bf16(
          false, a.v, false, bb2.v, (short)0, acc[j], false, false);
    }
  }
  __syncthreads();  // rden visible to all lanes

#pragma unroll
  for (int j = 0; j < 4; ++j) {
#pragma unroll
    for (int r = 0; r < 8; ++r) {
      const int row = wave * 16 + hi * 8 + r;   // 0..127
      const int e   = j * 16 + lane_r;          // 0..63
      const float o = acc[j][r] * rden[row];
      const size_t m = (size_t)b * SEQ + (s0 + row);
      att[m * HIDDEN + h * HD + e] = (bf16_t)o;
    }
  }
}

// ---------------------------------------------------------------------------
// Launch
// ---------------------------------------------------------------------------
extern "C" void kernel_launch(void* const* d_in, const int* in_sizes, int n_in,
                              void* d_out, int out_size, void* d_ws,
                              size_t ws_size, hipStream_t stream) {
  const float* X  = (const float*)d_in[0];   // [4, 8192, 1024]
  const float* Wq = (const float*)d_in[1];   // [1024, 1024]
  const float* bq = (const float*)d_in[2];   // [1024]
  // d_in[3..6] = Wk, bk, Wv, bv: dead w.r.t. the reference's return value
  const float* Wo = (const float*)d_in[7];   // [1024, 1024]
  const float* S0 = (const float*)d_in[8];   // [4, 16, 64, 64]
  const float* z0 = (const float*)d_in[9];   // [4, 16, 64]
  float* out = (float*)d_out;                // [4, 8192, 1024] fp32

  // Workspace (bf16): Wq | Wo | S0^T | Xb | qf(head-major) | att(merged)
  bf16_t* Wqb = (bf16_t*)d_ws;
  bf16_t* Wob = Wqb + (size_t)HIDDEN * HIDDEN;
  bf16_t* S0t = Wob + (size_t)HIDDEN * HIDDEN;
  bf16_t* Xb  = S0t + (size_t)BATCH * NHEADS * HD * HD;
  bf16_t* qf  = Xb + (size_t)MROWS * HIDDEN;
  bf16_t* att = qf + (size_t)MROWS * HIDDEN;

  const int nw = HIDDEN * HIDDEN;
  lma_cvt_f32_bf16_kernel<<<nw / 256, 256, 0, stream>>>(Wq, Wqb, nw);
  lma_cvt_f32_bf16_kernel<<<nw / 256, 256, 0, stream>>>(Wo, Wob, nw);
  lma_s0_transpose_kernel<<<BATCH * NHEADS, 256, 0, stream>>>(S0, S0t);
  lma_cvt_x_kernel<<<((size_t)MROWS * HIDDEN) / (256 * 8), 256, 0, stream>>>(
      X, Xb);

  dim3 ggrid(HIDDEN / 128, MROWS / 128);  // (8, 256)
  lma_gemm_kernel<0><<<ggrid, 256, 0, stream>>>(Xb, Wqb, bq, (void*)qf);
  lma_retrieve_kernel<<<dim3(SEQ / 128, BATCH * NHEADS), 256, 0, stream>>>(
      qf, S0t, z0, att);
  lma_gemm_kernel<1><<<ggrid, 256, 0, stream>>>(att, Wob, nullptr, (void*)out);

  (void)in_sizes; (void)n_in; (void)out_size; (void)ws_size;
}
